// ModifiedChebyLayer_7404523619230
// MI455X (gfx1250) — compile-verified, matrix operand order checked by reading, not run on hardware
//
#include <hip/hip_runtime.h>
#include <cstdint>

// ---------------------------------------------------------------------------
// ChebyKAN layer as an F16-WMMA GEMM:  out(32768,256) = cheb(x) @ W^T
//   cheb(x): (32768, 2304), K index k = in*9 + d, value T_d(tanh(x[.,in]))
//   W      : (256, 2304) = (c_basis * c_act[...,None]).reshape
// M=32768, K=2304, N=256  -> compute-bound; v_wmma_f32_16x16x32_f16.
// Tiling: WG = 256 thr (8 waves), owns 32 rows x 256 cols.
//   Each wave: 2 M-tiles x 2 N-tiles (32x32 output); per K-step
//   2 A frags (LDS) + 2 B frags (global, saddr+imm) feed 4 WMMAs,
//   software-pipelined one K-step ahead.
// ---------------------------------------------------------------------------

typedef _Float16 v8h  __attribute__((ext_vector_type(8)));
typedef _Float16 v16h __attribute__((ext_vector_type(16)));
typedef float    v8f  __attribute__((ext_vector_type(8)));

#define BATCH   32768
#define N_IN    256
#define N_OUT   256
#define DEG     8
#define KDIM    2304          // N_IN * (DEG+1)
#define KT_ALL  72            // KDIM / 32
#define ROWS    32            // batch rows per workgroup (2 M-tiles)
#define CHUNK_IN 64           // inputs per K-chunk
#define CHUNK_K  576          // CHUNK_IN * 9
#define KT_CHUNK 18           // CHUNK_K / 32
#define NCHUNK   4            // N_IN / CHUNK_IN
#define ROW_STRIDE 584        // f16 elems: 576 + 8 pad (16B-aligned rows, bank spread)
#define NT_DELTA (8 * KT_ALL * 32 * 16 * 2)   // byte delta between N-tile nt and nt+8 = 589824

// ---------------------------------------------------------------------------
// Kernel 1: fuse c_basis*c_act -> f16, pre-swizzled into WMMA B-fragment order.
// Fragment element j of lane l in K-tile kt / N-tile nt corresponds to
//   k_local = (j>=8 ? 16:0) + (l>=16 ? 8:0) + (j&7),  n = nt*16 + (l&15)
// (mirror of the documented 16-bit 16x32 A layout). Linear address:
//   ((nt*72 + kt)*32 + l)*16 + j
// ---------------------------------------------------------------------------
__global__ void cheb_pack_w(const float* __restrict__ c_basis,
                            const float* __restrict__ c_act,
                            _Float16* __restrict__ wbuf)
{
    int idx = blockIdx.x * blockDim.x + threadIdx.x;     // 0 .. N_OUT*KDIM-1
    if (idx >= N_OUT * KDIM) return;
    int n = idx / KDIM;
    int k = idx - n * KDIM;
    int in = k / 9;                                      // k = in*9 + d
    float w = c_basis[n * KDIM + k] * c_act[n * N_IN + in];

    int nt = n >> 4;
    int lm = n & 15;
    int kt = k >> 5;
    int kl = k & 31;
    int lane = lm + ((kl & 8) ? 16 : 0);
    int j    = ((kl & 16) ? 8 : 0) + (kl & 7);
    wbuf[(((size_t)nt * KT_ALL + kt) * 32 + lane) * 16 + j] = (_Float16)w;
}

// ---------------------------------------------------------------------------
// Kernel 2: fused Chebyshev-basis generation + WMMA GEMM.
// ---------------------------------------------------------------------------
__global__ __launch_bounds__(256, 1)
void cheb_gemm(const float* __restrict__ x,
               const _Float16* __restrict__ wbuf,
               float* __restrict__ out)
{
    __shared__ _Float16 lds[ROWS * ROW_STRIDE];          // 36.5 KB

    const int tid     = threadIdx.x;
    const int wave    = tid >> 5;                        // 0..7
    const int lane    = tid & 31;
    const int rowbase = blockIdx.x * ROWS;

    const int nt0 = wave;                                // N-tiles nt0, nt0+8
    v8f acc00 = {};                                      // (M-tile 0, nt0)
    v8f acc01 = {};                                      // (M-tile 0, nt0+8)
    v8f acc10 = {};                                      // (M-tile 1, nt0)
    v8f acc11 = {};                                      // (M-tile 1, nt0+8)

    // Per-lane A-fragment sources in LDS (row-major cheb slab for this chunk)
    const int ahi = (lane >> 4) ? 8 : 0;                 // lanes 16-31: +8 in K
    const _Float16* aptr0 = &lds[(lane & 15) * ROW_STRIDE + ahi];         // rows 0-15
    const _Float16* aptr1 = &lds[(16 + (lane & 15)) * ROW_STRIDE + ahi];  // rows 16-31

    // B stream: one 32-bit byte offset against saddr=wbuf; nt+8 stream and the
    // per-K-tile advance live in 24-bit signed instruction offsets.
    const char* wb = (const char*)wbuf;
    const unsigned wlane = ((unsigned)(nt0 * KT_ALL) * 32u + (unsigned)lane) * 32u; // bytes

    // Generation role: thread handles 8 consecutive inputs of one row.
    const int grow = tid >> 3;                           // 0..31
    const int giq  = tid & 7;                            // 8 octets * 8 = 64 inputs

    for (int chunk = 0; chunk < NCHUNK; ++chunk) {
        __syncthreads();   // previous chunk's WMMA reads done before overwrite

        // ---- generate T_0..T_8(tanh(x)) for 8 inputs -> 72 f16 -> LDS ----
        {
            const int in0 = chunk * CHUNK_IN + giq * 8;
            const float* xr = &x[(size_t)(rowbase + grow) * N_IN + in0];
            const float4 xv0 = *(const float4*)(xr);
            const float4 xv1 = *(const float4*)(xr + 4);

            union { _Float16 h[72]; unsigned long long q[18]; } vals;
            const float tv[8] = { tanhf(xv0.x), tanhf(xv0.y), tanhf(xv0.z), tanhf(xv0.w),
                                  tanhf(xv1.x), tanhf(xv1.y), tanhf(xv1.z), tanhf(xv1.w) };
#pragma unroll
            for (int e = 0; e < 8; ++e) {
                float t  = tv[e];
                float p0 = 1.0f, p1 = t;
                vals.h[e * 9 + 0] = (_Float16)1.0f;
                vals.h[e * 9 + 1] = (_Float16)t;
#pragma unroll
                for (int d = 2; d <= DEG; ++d) {
                    float pn = 2.0f * t * p1 - p0;
                    vals.h[e * 9 + d] = (_Float16)pn;
                    p0 = p1; p1 = pn;
                }
            }
            // 72 f16 = 144 bytes, 8B-aligned dest -> 18x ds_store_b64
            unsigned long long* dst =
                (unsigned long long*)&lds[grow * ROW_STRIDE + giq * 72];
#pragma unroll
            for (int i = 0; i < 18; ++i) dst[i] = vals.q[i];
        }

        __syncthreads();   // slab visible to all waves

        // byte offset of this chunk's first K-tile within the nt0 B stream
        const unsigned boff = wlane + (unsigned)(chunk * KT_CHUNK) * 1024u;
        const char* w0 = wb + boff;                      // + kt*1024            (imm)
                                                         // + NT_DELTA + kt*1024 (imm)

        // ---- 18 K-tiles, software-pipelined one step ahead ----------------
        v16h a0c, a1c, b0c, b1c;
        {
            v8h l0 = *(const v8h*)(aptr0);
            v8h h0 = *(const v8h*)(aptr0 + 16);
            v8h l1 = *(const v8h*)(aptr1);
            v8h h1 = *(const v8h*)(aptr1 + 16);
            a0c = __builtin_shufflevector(l0, h0, 0,1,2,3,4,5,6,7,8,9,10,11,12,13,14,15);
            a1c = __builtin_shufflevector(l1, h1, 0,1,2,3,4,5,6,7,8,9,10,11,12,13,14,15);
            b0c = *(const v16h*)(w0);
            b1c = *(const v16h*)(w0 + NT_DELTA);
        }
#pragma unroll
        for (int kt = 0; kt < KT_CHUNK - 1; ++kt) {
            // prefetch K-step kt+1 into fresh registers
            v8h l0 = *(const v8h*)(aptr0 + (kt + 1) * 32);
            v8h h0 = *(const v8h*)(aptr0 + (kt + 1) * 32 + 16);
            v8h l1 = *(const v8h*)(aptr1 + (kt + 1) * 32);
            v8h h1 = *(const v8h*)(aptr1 + (kt + 1) * 32 + 16);
            v16h a0n = __builtin_shufflevector(l0, h0, 0,1,2,3,4,5,6,7,8,9,10,11,12,13,14,15);
            v16h a1n = __builtin_shufflevector(l1, h1, 0,1,2,3,4,5,6,7,8,9,10,11,12,13,14,15);
            v16h b0n = *(const v16h*)(w0 + (kt + 1) * 1024);
            v16h b1n = *(const v16h*)(w0 + NT_DELTA + (kt + 1) * 1024);

            acc00 = __builtin_amdgcn_wmma_f32_16x16x32_f16(
                        false, a0c, false, b0c, (short)0, acc00, false, false);
            acc10 = __builtin_amdgcn_wmma_f32_16x16x32_f16(
                        false, a1c, false, b0c, (short)0, acc10, false, false);
            acc01 = __builtin_amdgcn_wmma_f32_16x16x32_f16(
                        false, a0c, false, b1c, (short)0, acc01, false, false);
            acc11 = __builtin_amdgcn_wmma_f32_16x16x32_f16(
                        false, a1c, false, b1c, (short)0, acc11, false, false);

            a0c = a0n; a1c = a1n; b0c = b0n; b1c = b1n;
        }
        // drain last K-step of the chunk
        acc00 = __builtin_amdgcn_wmma_f32_16x16x32_f16(
                    false, a0c, false, b0c, (short)0, acc00, false, false);
        acc10 = __builtin_amdgcn_wmma_f32_16x16x32_f16(
                    false, a1c, false, b0c, (short)0, acc10, false, false);
        acc01 = __builtin_amdgcn_wmma_f32_16x16x32_f16(
                    false, a0c, false, b1c, (short)0, acc01, false, false);
        acc11 = __builtin_amdgcn_wmma_f32_16x16x32_f16(
                    false, a1c, false, b1c, (short)0, acc11, false, false);
    }

    // ---- store: C/D layout -> VGPR r holds (M = r + 8*(lane>=16), N = lane&15)
    const int orow = rowbase + ((lane >> 4) << 3);
    const int col  = nt0 * 16 + (lane & 15);
#pragma unroll
    for (int r = 0; r < 8; ++r) {
        out[(size_t)(orow + r) * N_OUT + col]             = acc00[r];
        out[(size_t)(orow + r) * N_OUT + col + 128]       = acc01[r];
        out[(size_t)(orow + 16 + r) * N_OUT + col]        = acc10[r];
        out[(size_t)(orow + 16 + r) * N_OUT + col + 128]  = acc11[r];
    }
}

// ---------------------------------------------------------------------------
extern "C" void kernel_launch(void* const* d_in, const int* in_sizes, int n_in,
                              void* d_out, int out_size, void* d_ws, size_t ws_size,
                              hipStream_t stream)
{
    const float* x       = (const float*)d_in[0];   // (32768, 256) f32
    const float* c_basis = (const float*)d_in[1];   // (256, 256, 9) f32
    const float* c_act   = (const float*)d_in[2];   // (256, 256) f32
    float*       out     = (float*)d_out;           // (32768, 256) f32
    _Float16*    wpacked = (_Float16*)d_ws;         // 256*2304 f16 = 1.18 MB

    // 1) pack + swizzle weights into WMMA B-fragment order
    cheb_pack_w<<<(N_OUT * KDIM + 255) / 256, 256, 0, stream>>>(c_basis, c_act, wpacked);

    // 2) fused cheb-basis + GEMM
    cheb_gemm<<<BATCH / ROWS, 256, 0, stream>>>(x, wpacked, out);
}
